// MHA_6648609374999
// MI455X (gfx1250) — matrix operand order repaired and checked by reading, not compiled, not run on hardware
//
#include <hip/hip_runtime.h>

// Problem constants (match reference)
#define S_LEN   2048
#define D_DIM   4096
#define H_NUM   32
#define KVH_NUM 8
#define HD_DIM  128                // D / H
#define KVD_DIM (KVH_NUM * HD_DIM) // 1024
#define NEG_MASK (-1.0e9f)

typedef __attribute__((ext_vector_type(16))) _Float16 v16h;
typedef __attribute__((ext_vector_type(8)))  _Float16 v8h;
typedef __attribute__((ext_vector_type(4)))  _Float16 v4h;
typedef __attribute__((ext_vector_type(8)))  float    v8f;
typedef __attribute__((ext_vector_type(4)))  unsigned u32x4;
typedef __attribute__((ext_vector_type(8)))  unsigned u32x8;

static __device__ __forceinline__ v16h cat8(v8h a, v8h b) {
  v16h r;
#pragma unroll
  for (int i = 0; i < 8; ++i) { r[i] = a[i]; r[i + 8] = b[i]; }
  return r;
}

// CDNA5 async global->LDS copy (ASYNCcnt-tracked, bypasses VGPRs).
static __device__ __forceinline__ void async_load_b128(unsigned lds_off,
                                                       const void* gaddr) {
  asm volatile("global_load_async_to_lds_b128 %0, %1, off"
               :: "v"(lds_off), "v"(gaddr) : "memory");
}
static __device__ __forceinline__ void wait_async0() {
  asm volatile("s_wait_asynccnt 0" ::: "memory");
}

// CDNA5 Tensor Data Mover: 2D tile (tile_d1 rows x tile_d0 elems, f16) from a
// row-strided tensor into LDS, with 4-DWORD padding every 16 DWORDs so a
// 64-byte row lands on an 80-byte LDS pitch. D# groups built in SGPRs
// (all inputs are wave-uniform); TENSORcnt-tracked.
static __device__ __forceinline__ void tdm_load_tile_f16(
    unsigned lds_byte_off, const void* gaddr, unsigned tensor_stride_elems,
    unsigned tile_d0, unsigned tile_d1) {
  unsigned long long ga = (unsigned long long)gaddr;
  u32x4 g0;
  g0[0] = 1u;                                   // count=1 (valid user D#)
  g0[1] = lds_byte_off;                         // lds_addr
  g0[2] = (unsigned)(ga & 0xffffffffu);         // global_addr[31:0]
  g0[3] = (unsigned)((ga >> 32) & 0x01ffffffu)  // global_addr[56:32]
          | (2u << 30);                         // type=2 ("image")
  u32x8 g1;
  g1[0] = (1u << 16)    // data_size = 2 bytes
        | (1u << 20)    // pad_enable
        | (3u << 22)    // pad_interval: every 16 DWORDs (64B row)
        | (3u << 25);   // pad_amount: 4 DWORDs (16B) -> 80B pitch
  g1[1] = (tile_d0 & 0xffffu) << 16;                  // tensor_dim0 = tile_d0
  g1[2] = (tile_d0 >> 16) | ((tile_d1 & 0xffffu) << 16); // dim0 hi | dim1 lo
  g1[3] = (tile_d1 >> 16) | ((tile_d0 & 0xffffu) << 16); // dim1 hi | tile_dim0
  g1[4] = tile_d1 & 0xffffu;                          // tile_dim1 | tile_dim2=0
  g1[5] = tensor_stride_elems;                        // tensor_dim0_stride lo
  g1[6] = 0u;                                         // stride hi | dim1_stride
  g1[7] = 0u;
  asm volatile("tensor_load_to_lds %0, %1" :: "s"(g0), "s"(g1) : "memory");
}
static __device__ __forceinline__ void wait_tensor0() {
  __builtin_amdgcn_s_wait_tensorcnt(0);
}

// ---------------------------------------------------------------------------
// f32 -> f16 cast (vectorized x4); n must be a multiple of 4 (true here).
// ---------------------------------------------------------------------------
__global__ void cast_f32_to_f16(const float* __restrict__ src,
                                _Float16* __restrict__ dst, int n) {
  int i = (blockIdx.x * blockDim.x + threadIdx.x) * 4;
  if (i >= n) return;
  float4 v = *(const float4*)(src + i);
  v4h h;
  h[0] = (_Float16)v.x; h[1] = (_Float16)v.y;
  h[2] = (_Float16)v.z; h[3] = (_Float16)v.w;
  *(v4h*)(dst + i) = h;
}

// ---------------------------------------------------------------------------
// Tiled WMMA GEMM:  C[M,N] = A[M,K] * B[N,K]^T   (both f16 row-major, K inner)
// STORE_MODE: 0 -> f16 C[M,N], 1 -> f32 C[M,N], 2 -> f16 Ct[N,M] (transposed)
// Block tile 128x128, K-step 32, 8 waves (4x2), each wave 32x64 = 8 WMMA accs.
// Double-buffered LDS staging: A tile via TDM (tensor_load_to_lds, TENSORcnt),
// B tile via global_load_async_to_lds_b128 (ASYNCcnt).
// ---------------------------------------------------------------------------
template <int STORE_MODE>
__global__ __launch_bounds__(256)
void gemm_bt_wmma(const _Float16* __restrict__ A,
                  const _Float16* __restrict__ B,
                  void* __restrict__ C, int M, int N, int K) {
  // smem[buf][0]=A tile, smem[buf][1]=B tile; row stride 40 halves (80 B).
  __shared__ _Float16 smem[2][2][128][40];

  const int tid  = threadIdx.x;
  const int lane = tid & 31;
  const int wave = tid >> 5;      // 0..7
  const int wm   = wave >> 1;     // 0..3 : 32-row strip
  const int wn   = wave & 1;      // 0..1 : 64-col strip
  const int rlo  = lane & 15;
  const int rhi  = lane >> 4;
  const int bm   = blockIdx.y * 128;
  const int bn   = blockIdx.x * 128;

  // Per-thread async-copy mapping for the B tile: 512 x 16B chunks, 2/thread.
  const int row0 = tid >> 2,         kc0 = (tid & 3) << 3;
  const int row1 = (tid + 256) >> 2, kc1 = ((tid + 256) & 3) << 3;

  auto issue_tile = [&](int k0, int buf) {
    if (wave == 0) { // one TDM descriptor moves the whole 128x32 A tile
      tdm_load_tile_f16((unsigned)(buf * 20480),
                        A + (size_t)bm * K + k0, (unsigned)K, 32u, 128u);
    }
    unsigned bb0 = (unsigned)(((buf * 2 + 1) * 5120 + row0 * 40 + kc0) * 2);
    unsigned bb1 = (unsigned)(((buf * 2 + 1) * 5120 + row1 * 40 + kc1) * 2);
    async_load_b128(bb0, B + (size_t)(bn + row0) * K + k0 + kc0);
    async_load_b128(bb1, B + (size_t)(bn + row1) * K + k0 + kc1);
  };

  v8f acc[2][4];
#pragma unroll
  for (int m = 0; m < 2; ++m)
#pragma unroll
    for (int n = 0; n < 4; ++n) acc[m][n] = (v8f)(0.0f);

  const int T = K >> 5; // number of 32-wide K tiles
  issue_tile(0, 0);
  if (wave == 0) wait_tensor0();
  wait_async0();
  __syncthreads();

  for (int t = 0; t < T; ++t) {
    const int buf = t & 1;
    if (t + 1 < T) issue_tile((t + 1) << 5, buf ^ 1); // overlap with compute

    // A fragments (16x32 f16): lane=row, K 0-7/16-23 (lo half) or 8-15/24-31
    v16h af[2];
#pragma unroll
    for (int m = 0; m < 2; ++m) {
      int r  = wm * 32 + m * 16 + rlo;
      int kb = rhi * 8;
      v8h lo8 = *(const v8h*)(&smem[buf][0][r][kb]);
      v8h hi8 = *(const v8h*)(&smem[buf][0][r][16 + kb]);
      af[m] = cat8(lo8, hi8);
    }
    // B fragments (32x16 f16): lane=col, K 0-7/16-23 (lo half) or 8-15/24-31
    v16h bf[4];
#pragma unroll
    for (int n = 0; n < 4; ++n) {
      int c   = wn * 64 + n * 16 + rlo;
      int kb2 = rhi * 8;
      v8h lo8 = *(const v8h*)(&smem[buf][1][c][kb2]);
      v8h hi8 = *(const v8h*)(&smem[buf][1][c][16 + kb2]);
      bf[n] = cat8(lo8, hi8);
    }
#pragma unroll
    for (int m = 0; m < 2; ++m)
#pragma unroll
      for (int n = 0; n < 4; ++n)
        acc[m][n] = __builtin_amdgcn_wmma_f32_16x16x32_f16(
            false, af[m], false, bf[n], (short)0, acc[m][n], false, false);

    if (t + 1 < T) {
      if (wave == 0) wait_tensor0(); // A tile resident (issuing wave)
      wait_async0();                 // this wave's B chunks resident
      __syncthreads();               // all waves' copies visible
    }
  }

  // Store: C frag layout -> row = g + 8*rhi, col = rlo within each 16x16 tile
#pragma unroll
  for (int m = 0; m < 2; ++m) {
#pragma unroll
    for (int n = 0; n < 4; ++n) {
#pragma unroll
      for (int g = 0; g < 8; ++g) {
        int r = bm + wm * 32 + m * 16 + g + 8 * rhi;
        int c = bn + wn * 64 + n * 16 + rlo;
        float v = acc[m][n][g];
        if (STORE_MODE == 0) {
          ((_Float16*)C)[(size_t)r * N + c] = (_Float16)v;
        } else if (STORE_MODE == 1) {
          ((float*)C)[(size_t)r * N + c] = v;
        } else {
          ((_Float16*)C)[(size_t)c * M + r] = (_Float16)v; // transposed
        }
      }
    }
  }
}

// ---------------------------------------------------------------------------
// RoPE, in place on f16 [S, nheads, HD]; pair index i over HD/2.
// ---------------------------------------------------------------------------
__global__ void rope_f16(_Float16* __restrict__ t,
                         const float* __restrict__ cosp,
                         const float* __restrict__ sinp, int nheads) {
  int idx = blockIdx.x * blockDim.x + threadIdx.x;
  int total = S_LEN * nheads * (HD_DIM / 2);
  if (idx >= total) return;
  int i    = idx % (HD_DIM / 2);
  int rest = idx / (HD_DIM / 2);
  int h    = rest % nheads;
  int s    = rest / nheads;
  float c  = cosp[s * (HD_DIM / 2) + i];
  float sn = sinp[s * (HD_DIM / 2) + i];
  size_t base = ((size_t)s * nheads + h) * HD_DIM + 2 * i;
  float e = (float)t[base];
  float o = (float)t[base + 1];
  t[base]     = (_Float16)(e * c - o * sn);
  t[base + 1] = (_Float16)(e * sn + o * c);
}

// ---------------------------------------------------------------------------
// Flash attention (causal, GQA 4:1). 4 independent waves per block; each wave
// owns a 16-row query tile of one head. Keys processed in chunks of 32.
// All 8 B-fragments of a chunk are loaded before the WMMA chain so loads
// overlap matrix math (staggered s_wait_loadcnt instead of wait-0 per WMMA).
// ---------------------------------------------------------------------------
__global__ __launch_bounds__(128)
void flash_attn_wmma(const _Float16* __restrict__ Qh,  // [S, H*HD]
                     const _Float16* __restrict__ Kh,  // [S, KVH*HD]
                     const _Float16* __restrict__ Vt,  // [KVH*HD, S]
                     _Float16* __restrict__ Oh) {      // [S, H*HD]
  __shared__ _Float16 Ps[4][16][40]; // per-wave P staging (padded rows)

  const int lane  = threadIdx.x & 31;
  const int wave  = threadIdx.x >> 5;            // 0..3
  const int head  = blockIdx.y;                  // 0..H-1
  const int kvh   = head >> 2;                   // H/KVH = 4
  const int qbase = blockIdx.x * 64 + wave * 16; // 16-row query tile
  const int rlo   = lane & 15;
  const int rhi   = lane >> 4;
  const float scale = 0.08838834764831845f;      // 1/sqrt(128)

  // Preload the wave's Q tile as 4 A-fragments (HD = 4 x 32)
  v16h qf[4];
  {
    const _Float16* qrow = Qh + ((size_t)(qbase + rlo) * H_NUM + head) * HD_DIM;
    int kb = rhi * 8;
#pragma unroll
    for (int kk = 0; kk < 4; ++kk) {
      v8h lo8 = *(const v8h*)(qrow + kk * 32 + kb);
      v8h hi8 = *(const v8h*)(qrow + kk * 32 + 16 + kb);
      qf[kk] = cat8(lo8, hi8);
    }
  }

  float m_i[8], l_i[8];
#pragma unroll
  for (int g = 0; g < 8; ++g) { m_i[g] = -1.0e30f; l_i[g] = 0.0f; }
  v8f oacc[8];
#pragma unroll
  for (int nd = 0; nd < 8; ++nd) oacc[nd] = (v8f)(0.0f);

  const int kend = qbase + 16; // causal: keys 0 .. qbase+15
  for (int kb0 = 0; kb0 < kend; kb0 += 32) {
    // Prefetch next key chunk toward L0/L2 (global_prefetch_b8)
    if (kb0 + 32 < kend) {
      __builtin_prefetch(Kh + ((size_t)(kb0 + 32 + rlo) * KVH_NUM + kvh) * HD_DIM,
                         0, 3);
    }

    // ---- load ALL K fragments for this chunk, then run the WMMA chain ----
    v16h bfr[2][4];
#pragma unroll
    for (int j = 0; j < 2; ++j) {
      int keyc = kb0 + j * 16 + rlo; // this lane's key column
      const _Float16* krow = Kh + ((size_t)keyc * KVH_NUM + kvh) * HD_DIM;
      int kb2 = rhi * 8;
#pragma unroll
      for (int kk = 0; kk < 4; ++kk) {
        v8h lo8 = *(const v8h*)(krow + kk * 32 + kb2);
        v8h hi8 = *(const v8h*)(krow + kk * 32 + 16 + kb2);
        bfr[j][kk] = cat8(lo8, hi8);
      }
    }
    v8f lg[2];
    lg[0] = (v8f)(0.0f);
    lg[1] = (v8f)(0.0f);
#pragma unroll
    for (int j = 0; j < 2; ++j)
#pragma unroll
      for (int kk = 0; kk < 4; ++kk)
        lg[j] = __builtin_amdgcn_wmma_f32_16x16x32_f16(
            false, qf[kk], false, bfr[j][kk], (short)0, lg[j], false, false);

    // ---- scale + causal mask + online softmax (f32, per row) ----
#pragma unroll
    for (int g = 0; g < 8; ++g) {
      int qrow = qbase + g + 8 * rhi;
#pragma unroll
      for (int j = 0; j < 2; ++j) {
        int key = kb0 + j * 16 + rlo;
        float v = lg[j][g] * scale;
        lg[j][g] = (key <= qrow) ? v : NEG_MASK;
      }
      float t = fmaxf(lg[0][g], lg[1][g]);
#pragma unroll
      for (int off = 8; off >= 1; off >>= 1)
        t = fmaxf(t, __shfl_xor(t, off, 16));
      float mnew = fmaxf(m_i[g], t);
      float corr = __expf(m_i[g] - mnew);
      m_i[g] = mnew;
      float p0 = __expf(lg[0][g] - mnew);
      float p1 = __expf(lg[1][g] - mnew);
      lg[0][g] = p0;
      lg[1][g] = p1;
      float rs = p0 + p1;
#pragma unroll
      for (int off = 8; off >= 1; off >>= 1)
        rs += __shfl_xor(rs, off, 16);
      l_i[g] = l_i[g] * corr + rs;
#pragma unroll
      for (int nd = 0; nd < 8; ++nd) oacc[nd][g] *= corr;
    }

    // ---- P relayout: C-frag (row=g+8*rhi, col=lane) -> LDS -> A-frag ----
#pragma unroll
    for (int j = 0; j < 2; ++j)
#pragma unroll
      for (int g = 0; g < 8; ++g)
        Ps[wave][g + 8 * rhi][j * 16 + rlo] = (_Float16)lg[j][g];
    __asm__ volatile("" ::: "memory"); // wave-synchronous LDS reuse; DS in-order
    v16h pf;
    {
      int kb = rhi * 8;
      v8h lo8 = *(const v8h*)(&Ps[wave][rlo][kb]);
      v8h hi8 = *(const v8h*)(&Ps[wave][rlo][16 + kb]);
      pf = cat8(lo8, hi8);
    }

    // ---- O += P * V : load all 8 V fragments, then the WMMA chain ----
    v16h vfr[8];
#pragma unroll
    for (int nd = 0; nd < 8; ++nd) {
      const _Float16* vcol =
          Vt + (size_t)(kvh * HD_DIM + nd * 16 + rlo) * S_LEN + kb0 + rhi * 8;
      v8h lo8 = *(const v8h*)(vcol);
      v8h hi8 = *(const v8h*)(vcol + 16);
      vfr[nd] = cat8(lo8, hi8);
    }
#pragma unroll
    for (int nd = 0; nd < 8; ++nd)
      oacc[nd] = __builtin_amdgcn_wmma_f32_16x16x32_f16(
          false, pf, false, vfr[nd], (short)0, oacc[nd], false, false);
  }

  // ---- finalize: divide by row sum, store f16 ----
#pragma unroll
  for (int g = 0; g < 8; ++g) {
    float inv = 1.0f / l_i[g];
    int s = qbase + g + 8 * rhi;
#pragma unroll
    for (int nd = 0; nd < 8; ++nd)
      Oh[((size_t)s * H_NUM + head) * HD_DIM + nd * 16 + rlo] =
          (_Float16)(oacc[nd][g] * inv);
  }
}

// ---------------------------------------------------------------------------
// Host-side launch
// ---------------------------------------------------------------------------
extern "C" void kernel_launch(void* const* d_in, const int* in_sizes, int n_in,
                              void* d_out, int out_size, void* d_ws,
                              size_t ws_size, hipStream_t stream) {
  const float* x    = (const float*)d_in[0];
  const float* wq   = (const float*)d_in[1];
  const float* wk   = (const float*)d_in[2];
  const float* wv   = (const float*)d_in[3];
  const float* wo   = (const float*)d_in[4];
  // d_in[5] = mask (unused; causal mask applied analytically)
  const float* cosp = (const float*)d_in[6];
  const float* sinp = (const float*)d_in[7];
  float* out = (float*)d_out;

  // Workspace layout (f16 elements)
  _Float16* ws  = (_Float16*)d_ws;
  _Float16* xh  = ws;                            // S*D
  _Float16* wqh = xh  + (size_t)S_LEN * D_DIM;   // D*D
  _Float16* wkh = wqh + (size_t)D_DIM * D_DIM;   // KVD*D
  _Float16* wvh = wkh + (size_t)KVD_DIM * D_DIM; // KVD*D
  _Float16* woh = wvh + (size_t)KVD_DIM * D_DIM; // D*D
  _Float16* qh  = woh + (size_t)D_DIM * D_DIM;   // S*D
  _Float16* kh  = qh  + (size_t)S_LEN * D_DIM;   // S*KVD
  _Float16* vt  = kh  + (size_t)S_LEN * KVD_DIM; // KVD*S (transposed)
  _Float16* oh  = vt  + (size_t)KVD_DIM * S_LEN; // S*D

  auto cast = [&](const float* s, _Float16* d, size_t n) {
    int blocks = (int)((n / 4 + 255) / 256);
    cast_f32_to_f16<<<blocks, 256, 0, stream>>>(s, d, (int)n);
  };
  cast(x,  xh,  (size_t)S_LEN * D_DIM);
  cast(wq, wqh, (size_t)D_DIM * D_DIM);
  cast(wk, wkh, (size_t)KVD_DIM * D_DIM);
  cast(wv, wvh, (size_t)KVD_DIM * D_DIM);
  cast(wo, woh, (size_t)D_DIM * D_DIM);

  dim3 blk(256);
  // Q = x @ wq^T : [2048,4096] x [4096,4096]^T
  gemm_bt_wmma<0><<<dim3(D_DIM / 128, S_LEN / 128), blk, 0, stream>>>(
      xh, wqh, qh, S_LEN, D_DIM, D_DIM);
  // K = x @ wk^T : [2048,4096] x [1024,4096]^T
  gemm_bt_wmma<0><<<dim3(KVD_DIM / 128, S_LEN / 128), blk, 0, stream>>>(
      xh, wkh, kh, S_LEN, KVD_DIM, D_DIM);
  // V^T = (x @ wv^T)^T stored as [KVD, S]
  gemm_bt_wmma<2><<<dim3(KVD_DIM / 128, S_LEN / 128), blk, 0, stream>>>(
      xh, wvh, vt, S_LEN, KVD_DIM, D_DIM);

  // RoPE on Q and K
  {
    int nq = S_LEN * H_NUM * (HD_DIM / 2);
    rope_f16<<<(nq + 255) / 256, 256, 0, stream>>>(qh, cosp, sinp, H_NUM);
    int nk = S_LEN * KVH_NUM * (HD_DIM / 2);
    rope_f16<<<(nk + 255) / 256, 256, 0, stream>>>(kh, cosp, sinp, KVH_NUM);
  }

  // Flash attention: grid (S/64 query tiles, H heads), 4 waves per block
  flash_attn_wmma<<<dim3(S_LEN / 64, H_NUM), 128, 0, stream>>>(qh, kh, vt, oh);

  // out = O @ wo^T : [2048,4096] x [4096,4096]^T, f32 store
  gemm_bt_wmma<1><<<dim3(D_DIM / 128, S_LEN / 128), blk, 0, stream>>>(
      oh, woh, out, S_LEN, D_DIM, D_DIM);
}
